// HeatConv_raw_att_46729244181085
// MI455X (gfx1250) — matrix operand order repaired and checked by reading, not compiled, not run on hardware
//
#include <hip/hip_runtime.h>
#include <hip/hip_bf16.h>
#include <math.h>

// Problem constants (match reference setup_inputs)
#define NN 32768   // nodes
#define EE 524288  // edges
#define BB 64      // graphs
#define LL 512     // nodes per graph
#define NHEAD 4

typedef __attribute__((ext_vector_type(16))) _Float16 v16h;
typedef __attribute__((ext_vector_type(8)))  float    v8f;

// ---------------------------------------------------------------------------
// WMMA fragment index maps (CDNA5 16x16x32 f16, wave32)
//  A (16xK, f16): lane L -> row m = L%16, half h = L/16.
//    element e: e<8  -> k = h*16 + e          (contiguous group 0)
//               e>=8 -> k = h*16 + e (contiguous group 1)
//  B (Kx16, f16): lane L -> col n = L%16, half h = L/16; element e -> k = h*16+e
//  C/D (16x16, f32): lane L -> col n = L%16; acc[r] -> row m = r + h*8
// ---------------------------------------------------------------------------
__device__ __forceinline__ float leaky02(float x) { return x > 0.f ? x : 0.2f * x; }
// order-preserving float<->uint map for atomicMax on floats
__device__ __forceinline__ unsigned fenc(float f) {
  unsigned u = __float_as_uint(f);
  return (u & 0x80000000u) ? ~u : (u | 0x80000000u);
}
__device__ __forceinline__ float fdec(unsigned u) {
  u = (u & 0x80000000u) ? (u & 0x7FFFFFFFu) : ~u;
  return __uint_as_float(u);
}

// load 8 contiguous f32 -> f16 fragment elements [base..base+7] (16B-aligned src)
__device__ __forceinline__ void set8(v16h& f, int base, const float* __restrict__ p) {
  float4 a = *reinterpret_cast<const float4*>(p);
  float4 b = *reinterpret_cast<const float4*>(p + 4);
  f[base + 0] = (_Float16)a.x; f[base + 1] = (_Float16)a.y;
  f[base + 2] = (_Float16)a.z; f[base + 3] = (_Float16)a.w;
  f[base + 4] = (_Float16)b.x; f[base + 5] = (_Float16)b.y;
  f[base + 6] = (_Float16)b.z; f[base + 7] = (_Float16)b.w;
}
__device__ __forceinline__ void zero8(v16h& f, int base) {
#pragma unroll
  for (int i = 0; i < 8; i++) f[base + i] = (_Float16)0.f;
}
__device__ __forceinline__ void set8_partial(v16h& f, int base, const float* __restrict__ p,
                                             int nvalid) {
#pragma unroll
  for (int i = 0; i < 8; i++) f[base + i] = (i < nvalid) ? (_Float16)p[i] : (_Float16)0.f;
}

// ---------------------------------------------------------------------------
// fills
// ---------------------------------------------------------------------------
__global__ void fill_f32_k(float* p, float v, int n) {
  int i = blockIdx.x * blockDim.x + threadIdx.x;
  if (i < n) p[i] = v;
}
__global__ void fill_u32_k(unsigned* p, unsigned v, int n) {
  int i = blockIdx.x * blockDim.x + threadIdx.x;
  if (i < n) p[i] = v;
}

// ---------------------------------------------------------------------------
// Generic WMMA GEMM:  C[M,Ncol] = epilogue( A[M,K] @ W[K,Ncol] + bias )
// one wave per 16x16 C tile. REQUIRES K % 32 == 0 (true for all call sites).
// Epilogue is a template parameter -> no runtime branches, batched R loads.
// ---------------------------------------------------------------------------
#define EP_NONE 0
#define EP_RES 1            // + R
#define EP_RELU 2           // relu(.)
#define EP_RES_RELUADD 3    // R + relu(.)

template <int EP>
__global__ void gemm_wmma_k(const float* __restrict__ A, int lda,
                            const float* __restrict__ W, int ldw,
                            const float* __restrict__ bias,
                            const float* __restrict__ R, int ldr,
                            float* __restrict__ C, int ldc,
                            int M, int K, int Ncol) {
  int gtid = blockIdx.x * blockDim.x + threadIdx.x;
  int wave = gtid >> 5;
  int lane = gtid & 31;
  int tilesN = Ncol >> 4;
  int tm = wave / tilesN;
  int tn = wave - tm * tilesN;
  if (tm * 16 >= M) return;
  int l16 = lane & 15, h = lane >> 4;
  int n = tn * 16 + l16;
  const float* arow = A + (size_t)(tm * 16 + l16) * lda;
  v8f acc = {};
  for (int kc = 0; kc < K; kc += 32) {
    v16h af, bf;
    // A: two contiguous 8-float groups, no bounds checks (K%32==0)
    set8(af, 0, arow + kc + h * 16);
    set8(af, 8, arow + kc + h * 16 + 8);
    // B: 16 strided dwords, unconditioned -> batched loads under one wait
    const float* wp = W + (size_t)(kc + h * 16) * ldw + n;
#pragma unroll
    for (int e = 0; e < 16; e++) bf[e] = (_Float16)wp[e * ldw];
    acc = __builtin_amdgcn_wmma_f32_16x16x32_f16(false, af, false, bf,
                                                 (short)0, acc, false, false);
  }
  float bs = bias[n];
  float res[8];
  if (EP == EP_RES || EP == EP_RES_RELUADD) {
#pragma unroll
    for (int r = 0; r < 8; r++) res[r] = R[(size_t)(tm * 16 + r + h * 8) * ldr + n];
  }
#pragma unroll
  for (int r = 0; r < 8; r++) {
    int m = tm * 16 + r + h * 8;
    float v = acc[r] + bs;
    if (EP == EP_RES)              v = v + res[r];
    else if (EP == EP_RELU)        v = fmaxf(v, 0.f);
    else if (EP == EP_RES_RELUADD) v = res[r] + fmaxf(v, 0.f);
    C[(size_t)m * ldc + n] = v;
  }
}

// ---------------------------------------------------------------------------
// HeteroLinear: Y[n,:] = X[n,:] @ hW[type(n)] + hb[type(n)]
// one wave -> 16 node rows; compute all 3 type GEMMs via WMMA, select per row
// ---------------------------------------------------------------------------
template <int KIN, int OUT>
__global__ void hetero_linear_k(const float* __restrict__ X,
                                const int* __restrict__ ntype,
                                const float* __restrict__ hW,   // [3,KIN,OUT]
                                const float* __restrict__ hb,   // [3,OUT]
                                float* __restrict__ Y) {
  int gtid = blockIdx.x * blockDim.x + threadIdx.x;
  int wave = gtid >> 5, lane = gtid & 31;
  if (wave * 16 >= NN) return;
  int l16 = lane & 15, h = lane >> 4;
  int row = wave * 16 + l16;
  static_assert(KIN <= 32, "single k-chunk");
  const float* xr = X + (size_t)row * KIN;
  v16h af;
  {
    int k0 = h * 16;         // group 0
    int nv = KIN - k0;
    if (nv >= 8)      set8(af, 0, xr + k0);
    else if (nv > 0)  set8_partial(af, 0, xr + k0, nv);
    else              zero8(af, 0);
    int k1 = h * 16 + 8;     // group 1
    int nv1 = KIN - k1;
    if (nv1 >= 8)     set8(af, 8, xr + k1);
    else if (nv1 > 0) set8_partial(af, 8, xr + k1, nv1);
    else              zero8(af, 8);
  }
  constexpr int NTN = OUT / 16;
  for (int t = 0; t < 3; t++) {
    v8f acc[NTN];
#pragma unroll
    for (int i = 0; i < NTN; i++) { v8f z = {}; acc[i] = z; }
#pragma unroll
    for (int nt = 0; nt < NTN; nt++) {
      v16h bf;
      int kb0 = h * 16;
      int nv = KIN - kb0;
      if (nv >= 16) {
        const float* wp = hW + (size_t)(t * KIN + kb0) * OUT + nt * 16 + l16;
#pragma unroll
        for (int e = 0; e < 16; e++) bf[e] = (_Float16)wp[e * OUT];
      } else if (nv > 0) {
        const float* wp = hW + (size_t)(t * KIN + kb0) * OUT + nt * 16 + l16;
#pragma unroll
        for (int e = 0; e < 16; e++)
          bf[e] = (e < nv) ? (_Float16)wp[e * OUT] : (_Float16)0.f;
      } else {
        zero8(bf, 0); zero8(bf, 8);
      }
      acc[nt] = __builtin_amdgcn_wmma_f32_16x16x32_f16(false, af, false, bf,
                                                       (short)0, acc[nt], false, false);
    }
#pragma unroll
    for (int r = 0; r < 8; r++) {
      int m = wave * 16 + r + h * 8;
      if (ntype[m] == t) {
#pragma unroll
        for (int nt = 0; nt < NTN; nt++)
          Y[m * OUT + nt * 16 + l16] = acc[nt][r] + hb[t * OUT + nt * 16 + l16];
      }
    }
  }
}

// ---------------------------------------------------------------------------
// per-edge ea = leaky(eattr @ eaW)   [E,16]
// ---------------------------------------------------------------------------
__global__ void edge_ea_k(const float* __restrict__ ef, const float* __restrict__ eaW,
                          float* __restrict__ ea) {
  int e = blockIdx.x * blockDim.x + threadIdx.x;
  if (e >= EE) return;
  float f[6];
#pragma unroll
  for (int k = 0; k < 6; k++) f[k] = ef[e * 6 + k];
#pragma unroll
  for (int j = 0; j < 16; j++) {
    float s = 0.f;
#pragma unroll
    for (int k = 0; k < 6; k++) s += f[k] * eaW[k * 16 + j];
    ea[e * 16 + j] = leaky02(s);
  }
}

// ---------------------------------------------------------------------------
// edge attention logits a = leaky([x_i,x_j,te,ea]@attW); atomic segment max
// vectorized float4 gathers (O % 4 == 0)
// ---------------------------------------------------------------------------
__global__ void edge_logits_k(const float* __restrict__ xh, int O,
                              const int* __restrict__ srcI, const int* __restrict__ dstI,
                              const int* __restrict__ etype, const float* __restrict__ ete,
                              const float* __restrict__ ea, const float* __restrict__ attW,
                              float* __restrict__ aout, unsigned* __restrict__ amax) {
  int e = blockIdx.x * blockDim.x + threadIdx.x;
  if (e >= EE) return;
  int s = srcI[e], d = dstI[e];
  float acc = 0.f;
  const float4* xi = reinterpret_cast<const float4*>(xh + (size_t)d * O);
  const float4* xj = reinterpret_cast<const float4*>(xh + (size_t)s * O);
  int O4 = O >> 2;
  for (int k = 0; k < O4; k++) {
    float4 a = xi[k];
    acc += a.x * attW[k * 4] + a.y * attW[k * 4 + 1] + a.z * attW[k * 4 + 2] + a.w * attW[k * 4 + 3];
  }
  for (int k = 0; k < O4; k++) {
    float4 a = xj[k];
    acc += a.x * attW[O + k * 4] + a.y * attW[O + k * 4 + 1] + a.z * attW[O + k * 4 + 2] +
           a.w * attW[O + k * 4 + 3];
  }
  const float* te = ete + etype[e] * 16;
#pragma unroll
  for (int k = 0; k < 16; k++) acc += leaky02(te[k]) * attW[2 * O + k];
  const float4* ev = reinterpret_cast<const float4*>(ea + (size_t)e * 16);
#pragma unroll
  for (int k = 0; k < 4; k++) {
    float4 a = ev[k];
    acc += a.x * attW[2 * O + 16 + k * 4] + a.y * attW[2 * O + 17 + k * 4] +
           a.z * attW[2 * O + 18 + k * 4] + a.w * attW[2 * O + 19 + k * 4];
  }
  float a = leaky02(acc);
  aout[e] = a;
  atomicMax(&amax[d], fenc(a));
}

// exp(a - amax[dst]) in place, segment sum of denominators
__global__ void edge_exp_k(float* __restrict__ aex, const int* __restrict__ dstI,
                           const unsigned* __restrict__ amax, float* __restrict__ den) {
  int e = blockIdx.x * blockDim.x + threadIdx.x;
  if (e >= EE) return;
  int d = dstI[e];
  float ex = __expf(aex[e] - fdec(amax[d]));
  aex[e] = ex;
  atomicAdd(&den[d], ex);
}

// ---------------------------------------------------------------------------
// edge messages: msg = [xh[src], ea] @ linW ; out[dst] += alpha * msg
// 16 edges per wave, WMMA over K=O+16 (zero-padded), atomic f32 scatter.
// O and KF are multiples of 8 (KF of 16) -> group-uniform validity, float4 loads.
// ---------------------------------------------------------------------------
template <int O>
__global__ void edge_msg_k(const float* __restrict__ xh, const float* __restrict__ ea,
                           const float* __restrict__ linW,
                           const int* __restrict__ srcI, const int* __restrict__ dstI,
                           const float* __restrict__ ex, const float* __restrict__ den,
                           float* __restrict__ out) {
  int gtid = blockIdx.x * blockDim.x + threadIdx.x;
  int wave = gtid >> 5, lane = gtid & 31;
  if (wave * 16 >= EE) return;
  int l16 = lane & 15, h = lane >> 4;
  constexpr int KF = O + 16;
  constexpr int KT = (KF + 31) / 32;
  constexpr int NTN = O / 16;
  int erow = wave * 16 + l16;
  int s = srcI[erow];
  const float* xr = xh + (size_t)s * O;
  const float* er = ea + (size_t)erow * 16;
  v8f acc[NTN];
#pragma unroll
  for (int i = 0; i < NTN; i++) { v8f z = {}; acc[i] = z; }
#pragma unroll
  for (int kc = 0; kc < KT; kc++) {
    v16h af;
    {
      int k0 = kc * 32 + h * 16;       // group 0 (8 contiguous k)
      if (k0 + 8 <= O)        set8(af, 0, xr + k0);
      else if (k0 >= O && k0 + 8 <= KF) set8(af, 0, er + (k0 - O));
      else                    zero8(af, 0);
      int k1 = kc * 32 + h * 16 + 8;   // group 1
      if (k1 + 8 <= O)        set8(af, 8, xr + k1);
      else if (k1 >= O && k1 + 8 <= KF) set8(af, 8, er + (k1 - O));
      else                    zero8(af, 8);
    }
#pragma unroll
    for (int nt = 0; nt < NTN; nt++) {
      v16h bf;
      int g0 = kc * 32 + h * 16;       // KF % 16 == 0 -> group fully valid or fully not
      if (g0 + 16 <= KF) {
        const float* wp = linW + (size_t)g0 * O + nt * 16 + l16;
#pragma unroll
        for (int e = 0; e < 16; e++) bf[e] = (_Float16)wp[e * O];
      } else {
        zero8(bf, 0); zero8(bf, 8);
      }
      acc[nt] = __builtin_amdgcn_wmma_f32_16x16x32_f16(false, af, false, bf,
                                                       (short)0, acc[nt], false, false);
    }
  }
#pragma unroll
  for (int r = 0; r < 8; r++) {
    int m = wave * 16 + r + h * 8;
    int d = dstI[m];
    float alpha = ex[m] / (den[d] + 1e-16f);
#pragma unroll
    for (int nt = 0; nt < NTN; nt++)
      atomicAdd(&out[d * O + nt * 16 + l16], acc[nt][r] * alpha);
  }
}

// ---------------------------------------------------------------------------
// batch norm stats + apply(ELU)
// ---------------------------------------------------------------------------
__global__ void bn_stats_k(const float* __restrict__ X, float* __restrict__ st, int O) {
  int idx = blockIdx.x * blockDim.x + threadIdx.x;
  if (idx >= O * 256) return;
  int j = idx % O, ch = idx / O;
  float s = 0.f, q = 0.f;
  for (int r = ch * 128; r < ch * 128 + 128; r++) {
    float v = X[r * O + j];
    s += v; q += v * v;
  }
  atomicAdd(&st[j], s);
  atomicAdd(&st[64 + j], q);
}
__global__ void bn_apply_k(float* __restrict__ X, const float* __restrict__ st,
                           const float* __restrict__ g, const float* __restrict__ b, int O) {
  int idx = blockIdx.x * blockDim.x + threadIdx.x;
  if (idx >= NN * O) return;
  int j = idx % O;
  float mu = st[j] * (1.f / NN);
  float var = st[64 + j] * (1.f / NN) - mu * mu;
  float v = (X[idx] - mu) * rsqrtf(var + 1e-5f) * g[j] + b[j];
  X[idx] = v > 0.f ? v : expm1f(v);  // ELU
}

// ---------------------------------------------------------------------------
// Fused SAB attention (flash style, per-wave: one (graph, head, 16-query) tile)
// qkv: [B*L, 192] (q|k|v), out: [B*L, 64] (heads concatenated)
// Computes S^T = K_chunk @ Q^T so the softmax tile is already in the lane
// layout required for the A operand of P @ V (no LDS transpose).
// ---------------------------------------------------------------------------
__global__ void sab_attention_k(const float* __restrict__ qkv, float* __restrict__ out) {
  int gtid = blockIdx.x * blockDim.x + threadIdx.x;
  int wave = gtid >> 5, lane = gtid & 31;
  const int nqt = LL / 16;  // 32
  if (wave >= BB * NHEAD * nqt) return;
  int qt = wave % nqt;
  int hh = (wave / nqt) % NHEAD;
  int b = wave / (nqt * NHEAD);
  int l16 = lane & 15, h = lane >> 4;

  // Q^T B-fragment (d x 16 queries): d = h*16+e -> h==0 lanes hold 16 contiguous d
  v16h qb;
  if (h == 0) {
    const float* qp = qkv + (size_t)(b * LL + qt * 16 + l16) * 192 + hh * 16;
    float4 a0 = *reinterpret_cast<const float4*>(qp);
    float4 a1 = *reinterpret_cast<const float4*>(qp + 4);
    float4 a2 = *reinterpret_cast<const float4*>(qp + 8);
    float4 a3 = *reinterpret_cast<const float4*>(qp + 12);
    qb[0] = (_Float16)(a0.x * 0.25f);  qb[1] = (_Float16)(a0.y * 0.25f);
    qb[2] = (_Float16)(a0.z * 0.25f);  qb[3] = (_Float16)(a0.w * 0.25f);
    qb[4] = (_Float16)(a1.x * 0.25f);  qb[5] = (_Float16)(a1.y * 0.25f);
    qb[6] = (_Float16)(a1.z * 0.25f);  qb[7] = (_Float16)(a1.w * 0.25f);
    qb[8] = (_Float16)(a2.x * 0.25f);  qb[9] = (_Float16)(a2.y * 0.25f);
    qb[10] = (_Float16)(a2.z * 0.25f); qb[11] = (_Float16)(a2.w * 0.25f);
    qb[12] = (_Float16)(a3.x * 0.25f); qb[13] = (_Float16)(a3.y * 0.25f);
    qb[14] = (_Float16)(a3.z * 0.25f); qb[15] = (_Float16)(a3.w * 0.25f);
  } else {
    zero8(qb, 0); zero8(qb, 8);
  }
  float mrun = -INFINITY, denom = 0.f;
  v8f oacc = {};
  for (int kc = 0; kc < LL / 16; kc++) {
    // K chunk as A (16 keys x d=16 padded to 32): d = h*16+e -> only h==0 valid
    v16h kf;
    if (h == 0) {
      const float* kp = qkv + (size_t)(b * LL + kc * 16 + l16) * 192 + 64 + hh * 16;
      set8(kf, 0, kp);
      set8(kf, 8, kp + 8);
    } else {
      zero8(kf, 0); zero8(kf, 8);  // d >= 16 pad
    }
    v8f st = {};
    st = __builtin_amdgcn_wmma_f32_16x16x32_f16(false, kf, false, qb,
                                                (short)0, st, false, false);
    // st: lane's query = l16, rows r -> key h*8+r
    float cmax = st[0];
#pragma unroll
    for (int r = 1; r < 8; r++) cmax = fmaxf(cmax, st[r]);
    cmax = fmaxf(cmax, __shfl_xor(cmax, 16, 32));
    float mnew = fmaxf(mrun, cmax);
    float corr = __expf(mrun - mnew);
    mrun = mnew;
    float p[8], psum = 0.f;
#pragma unroll
    for (int r = 0; r < 8; r++) { p[r] = __expf(st[r] - mnew); psum += p[r]; }
    psum += __shfl_xor(psum, 16, 32);
    denom = denom * corr + psum;
    // P as A fragment: lane row=query l16, elements e<8 are keys h*8+e
    v16h pf;
#pragma unroll
    for (int e = 0; e < 16; e++) pf[e] = (e < 8) ? (_Float16)p[e] : (_Float16)0.f;
    // V chunk as B (16 keys padded to 32 x 16 d): e<8 -> key h*8+e, e>=8 pad
    v16h vf;
    {
      const float* vp = qkv + (size_t)(b * LL + kc * 16 + h * 8) * 192 + 128 + hh * 16 + l16;
#pragma unroll
      for (int e = 0; e < 16; e++) vf[e] = (e < 8) ? (_Float16)vp[e * 192] : (_Float16)0.f;
    }
    // online rescale of accumulator rows (row m = query r + h*8)
#pragma unroll
    for (int r = 0; r < 8; r++) oacc[r] *= __shfl(corr, r + h * 8, 32);
    oacc = __builtin_amdgcn_wmma_f32_16x16x32_f16(false, pf, false, vf,
                                                  (short)0, oacc, false, false);
  }
#pragma unroll
  for (int r = 0; r < 8; r++) {
    int qq = r + h * 8;
    float dq = __shfl(denom, qq, 32);
    out[(size_t)(b * LL + qt * 16 + qq) * 64 + hh * 16 + l16] = oacc[r] / dq;
  }
}

// ---------------------------------------------------------------------------
// PMA attention: 1 query (shared seed) x 512 keys per (graph, head); one wave each
// kv: [B*L, 128] (k|v from y), out: [B, 64]
// ---------------------------------------------------------------------------
__global__ void pma_attention_k(const float* __restrict__ kv, const float* __restrict__ seed,
                                const float* __restrict__ qkvW, const float* __restrict__ qkvb,
                                float* __restrict__ out) {
  int gtid = blockIdx.x * blockDim.x + threadIdx.x;
  int wave = gtid >> 5, lane = gtid & 31;
  if (wave >= BB * NHEAD) return;
  int hh = wave & 3, b = wave >> 2;
  float q[16];
#pragma unroll
  for (int d = 0; d < 16; d++) {
    float s = qkvb[hh * 16 + d];
    for (int c = 0; c < 64; c++) s += seed[c] * qkvW[c * 192 + hh * 16 + d];
    q[d] = s * 0.25f;
  }
  float sc[16], mx = -INFINITY;
#pragma unroll
  for (int i = 0; i < 16; i++) {
    const float4* kp = reinterpret_cast<const float4*>(kv + (size_t)(b * LL + i * 32 + lane) * 128);
    float s = 0.f;
#pragma unroll
    for (int k4 = 0; k4 < 4; k4++) {
      float4 a = kp[k4];
      s += q[k4 * 4] * a.x + q[k4 * 4 + 1] * a.y + q[k4 * 4 + 2] * a.z + q[k4 * 4 + 3] * a.w;
    }
    sc[i] = s;
    mx = fmaxf(mx, s);
  }
#pragma unroll
  for (int off = 16; off >= 1; off >>= 1) mx = fmaxf(mx, __shfl_xor(mx, off, 32));
  float o[16];
#pragma unroll
  for (int d = 0; d < 16; d++) o[d] = 0.f;
  float dsum = 0.f;
#pragma unroll
  for (int i = 0; i < 16; i++) {
    float p = __expf(sc[i] - mx);
    dsum += p;
    const float4* vp =
        reinterpret_cast<const float4*>(kv + (size_t)(b * LL + i * 32 + lane) * 128 + 64);
#pragma unroll
    for (int k4 = 0; k4 < 4; k4++) {
      float4 a = vp[k4];
      o[k4 * 4] += p * a.x; o[k4 * 4 + 1] += p * a.y;
      o[k4 * 4 + 2] += p * a.z; o[k4 * 4 + 3] += p * a.w;
    }
  }
#pragma unroll
  for (int off = 16; off >= 1; off >>= 1) {
    dsum += __shfl_xor(dsum, off, 32);
#pragma unroll
    for (int d = 0; d < 16; d++) o[d] += __shfl_xor(o[d], off, 32);
  }
  if (lane == 0) {
#pragma unroll
    for (int d = 0; d < 16; d++) out[b * 64 + hh * 16 + d] = o[d] / dsum;
  }
}

// ---------------------------------------------------------------------------
// head: elu(fc1) -> fc2 -> log_softmax
// ---------------------------------------------------------------------------
__global__ void head_k(const float* __restrict__ X, const float* __restrict__ fc1W,
                       const float* __restrict__ fc1b, const float* __restrict__ fc2W,
                       const float* __restrict__ fc2b, float* __restrict__ out) {
  int b = threadIdx.x;
  if (b >= BB) return;
  float hf[16];
#pragma unroll
  for (int j = 0; j < 16; j++) {
    float s = fc1b[j];
    for (int k = 0; k < 64; k++) s += X[b * 64 + k] * fc1W[k * 16 + j];
    hf[j] = s > 0.f ? s : expm1f(s);
  }
  float l0 = fc2b[0], l1 = fc2b[1];
#pragma unroll
  for (int k = 0; k < 16; k++) { l0 += hf[k] * fc2W[k * 2]; l1 += hf[k] * fc2W[k * 2 + 1]; }
  float mx = fmaxf(l0, l1);
  float lse = mx + logf(__expf(l0 - mx) + __expf(l1 - mx));
  out[b * 2 + 0] = l0 - lse;
  out[b * 2 + 1] = l1 - lse;
}

// ---------------------------------------------------------------------------
// host-side orchestration
// ---------------------------------------------------------------------------
static inline int ceil_div(int a, int b) { return (a + b - 1) / b; }

static void launch_gemm(const float* A, int lda, const float* W, int ldw, const float* bias,
                        const float* R, int ldr, float* C, int ldc,
                        int M, int K, int Ncol, int ep, hipStream_t stream) {
  int waves = (M / 16) * (Ncol / 16);
  int blocks = ceil_div(waves * 32, 256);
  switch (ep) {
    case EP_RES:
      gemm_wmma_k<EP_RES><<<blocks, 256, 0, stream>>>(A, lda, W, ldw, bias, R, ldr, C, ldc, M, K, Ncol);
      break;
    case EP_RELU:
      gemm_wmma_k<EP_RELU><<<blocks, 256, 0, stream>>>(A, lda, W, ldw, bias, R, ldr, C, ldc, M, K, Ncol);
      break;
    case EP_RES_RELUADD:
      gemm_wmma_k<EP_RES_RELUADD><<<blocks, 256, 0, stream>>>(A, lda, W, ldw, bias, R, ldr, C, ldc, M, K, Ncol);
      break;
    default:
      gemm_wmma_k<EP_NONE><<<blocks, 256, 0, stream>>>(A, lda, W, ldw, bias, R, ldr, C, ldc, M, K, Ncol);
      break;
  }
}

template <int KIN, int O>
static void conv_layer(const float* Xin, void* const* d_in, int base,
                       const int* srcI, const int* dstI, const int* ntype,
                       const int* etype, const float* efeat,
                       float* xh, float* ea, float* aex, unsigned* amax, float* den,
                       float* stats, float* Yout, hipStream_t stream) {
  const float* hW   = (const float*)d_in[base + 0];
  const float* hb   = (const float*)d_in[base + 1];
  const float* ete  = (const float*)d_in[base + 2];
  const float* eaW  = (const float*)d_in[base + 3];
  const float* attW = (const float*)d_in[base + 4];
  const float* linW = (const float*)d_in[base + 5];
  const float* bng  = (const float*)d_in[base + 6];
  const float* bnb  = (const float*)d_in[base + 7];

  hetero_linear_k<KIN, O><<<ceil_div((NN / 16) * 32, 256), 256, 0, stream>>>(Xin, ntype, hW, hb, xh);
  edge_ea_k<<<ceil_div(EE, 256), 256, 0, stream>>>(efeat, eaW, ea);
  fill_u32_k<<<ceil_div(NN, 256), 256, 0, stream>>>(amax, 0x007FFFFFu, NN);  // enc(-inf)
  fill_f32_k<<<ceil_div(NN, 256), 256, 0, stream>>>(den, 0.f, NN);
  fill_f32_k<<<ceil_div(NN * O, 256), 256, 0, stream>>>(Yout, 0.f, NN * O);
  edge_logits_k<<<ceil_div(EE, 256), 256, 0, stream>>>(xh, O, srcI, dstI, etype, ete, ea,
                                                       attW, aex, amax);
  edge_exp_k<<<ceil_div(EE, 256), 256, 0, stream>>>(aex, dstI, amax, den);
  edge_msg_k<O><<<ceil_div((EE / 16) * 32, 256), 256, 0, stream>>>(xh, ea, linW, srcI, dstI,
                                                                   aex, den, Yout);
  fill_f32_k<<<1, 128, 0, stream>>>(stats, 0.f, 128);
  bn_stats_k<<<ceil_div(O * 256, 256), 256, 0, stream>>>(Yout, stats, O);
  bn_apply_k<<<ceil_div(NN * O, 256), 256, 0, stream>>>(Yout, stats, bng, bnb, O);
}

extern "C" void kernel_launch(void* const* d_in, const int* in_sizes, int n_in,
                              void* d_out, int out_size, void* d_ws, size_t ws_size,
                              hipStream_t stream) {
  (void)in_sizes; (void)n_in; (void)out_size; (void)ws_size;
  const float* x     = (const float*)d_in[0];
  const int*   srcI  = (const int*)d_in[1];
  const int*   dstI  = srcI + EE;
  const int*   ntype = (const int*)d_in[2];
  const int*   etype = (const int*)d_in[3];
  const float* efeat = (const float*)d_in[4];

  // workspace arena (floats), ~58 MB with explicit phase aliasing
  float* ws = (float*)d_ws;
  float* featA = ws;                       // [N,64]
  float* featB = featA + NN * 64;          // [N,64]
  float* aux   = featB + NN * 64;          // [N,64]  (xh in convs, attn_out later)
  float* aexb  = aux + NN * 64;            // [E]
  float* p0    = aexb + EE;
  unsigned* amax = (unsigned*)p0;          // [N] uint
  float* den   = p0 + NN;                  // [N]
  float* stats = den + NN;                 // [128]
  float* g0 = stats + 128;                 // 6 x [64,64]
  float* g1 = g0 + 4096;
  float* g2 = g1 + 4096;
  float* g3 = g2 + 4096;
  float* g4 = g3 + 4096;
  float* g5 = g4 + 4096;
  float* big = g5 + 4096;                  // [E,16] ea  <->  [N,192] qkv / [N,128] kv

  // ---- 3 HEATConv layers (ea lives in `big`, xh in `aux`) ----
  conv_layer<4, 16>(x,      d_in, 5,  srcI, dstI, ntype, etype, efeat, aux, big, aexb, amax, den, stats, featA, stream);
  conv_layer<16, 32>(featA, d_in, 13, srcI, dstI, ntype, etype, efeat, aux, big, aexb, amax, den, stats, featB, stream);
  conv_layer<32, 64>(featB, d_in, 21, srcI, dstI, ntype, etype, efeat, aux, big, aexb, amax, den, stats, featA, stream);
  // xd = featA

  const float* enc_qkvW = (const float*)d_in[29];
  const float* enc_qkvb = (const float*)d_in[30];
  const float* enc_oW   = (const float*)d_in[31];
  const float* enc_ob   = (const float*)d_in[32];
  const float* enc_lW   = (const float*)d_in[33];
  const float* enc_lb   = (const float*)d_in[34];
  const float* pma_qkvW = (const float*)d_in[35];
  const float* pma_qkvb = (const float*)d_in[36];
  const float* pma_oW   = (const float*)d_in[37];
  const float* pma_ob   = (const float*)d_in[38];
  const float* pma_lW   = (const float*)d_in[39];
  const float* pma_lb   = (const float*)d_in[40];
  const float* dec_qkvW = (const float*)d_in[41];
  const float* dec_qkvb = (const float*)d_in[42];
  const float* dec_oW   = (const float*)d_in[43];
  const float* dec_ob   = (const float*)d_in[44];
  const float* dec_lW   = (const float*)d_in[45];
  const float* dec_lb   = (const float*)d_in[46];
  const float* pma_pW   = (const float*)d_in[47];
  const float* pma_pb   = (const float*)d_in[48];
  const float* seed     = (const float*)d_in[49];
  const float* fc1W     = (const float*)d_in[50];
  const float* fc1b     = (const float*)d_in[51];
  const float* fc2W     = (const float*)d_in[52];
  const float* fc2b     = (const float*)d_in[53];

  // ---- encoder SAB ----
  launch_gemm(featA, 64, enc_qkvW, 192, enc_qkvb, nullptr, 0, big, 192, NN, 64, 192, EP_NONE, stream);
  sab_attention_k<<<ceil_div(BB * NHEAD * (LL / 16) * 32, 256), 256, 0, stream>>>(big, aux);
  launch_gemm(aux, 64, enc_oW, 64, enc_ob, featA, 64, featB, 64, NN, 64, 64, EP_RES, stream);          // out1
  launch_gemm(featB, 64, enc_lW, 64, enc_lb, featB, 64, aux, 64, NN, 64, 64, EP_RES_RELUADD, stream);  // out2

  // ---- PMA ----
  launch_gemm(aux, 64, pma_pW, 64, pma_pb, nullptr, 0, featA, 64, NN, 64, 64, EP_RELU, stream);        // y
  launch_gemm(featA, 64, pma_qkvW + 64, 192, pma_qkvb + 64, nullptr, 0, big, 128, NN, 64, 128, EP_NONE, stream); // k|v
  pma_attention_k<<<ceil_div(BB * NHEAD * 32, 256), 256, 0, stream>>>(big, seed, pma_qkvW, pma_qkvb, g0);
  launch_gemm(g0, 64, pma_oW, 64, pma_ob, seed, /*ldr=*/0, g1, 64, BB, 64, 64, EP_RES, stream);        // + seed residual
  launch_gemm(g1, 64, pma_lW, 64, pma_lb, g1, 64, g2, 64, BB, 64, 64, EP_RES_RELUADD, stream);

  // ---- decoder SAB (Lq=Lk=1 => attention output == v) ----
  launch_gemm(g2, 64, dec_qkvW + 128, 192, dec_qkvb + 128, nullptr, 0, g3, 64, BB, 64, 64, EP_NONE, stream); // v
  launch_gemm(g3, 64, dec_oW, 64, dec_ob, g2, 64, g4, 64, BB, 64, 64, EP_RES, stream);
  launch_gemm(g4, 64, dec_lW, 64, dec_lb, g4, 64, g5, 64, BB, 64, 64, EP_RES_RELUADD, stream);

  // ---- head ----
  head_k<<<1, 64, 0, stream>>>(g5, fc1W, fc1b, fc2W, fc2b, (float*)d_out);
}
